// GloAttnConv_82085414961439
// MI455X (gfx1250) — compile-verified, matrix operand order checked by reading, not run on hardware
//
#include <hip/hip_runtime.h>
#include <hip/hip_bf16.h>

// ---------------------------------------------------------------------------
// GloAttnConv: linear-attention graph conv, fused via v_wmma_f32_16x16x32_f16
// with double-buffered GLOBAL_LOAD_ASYNC_TO_LDS pipelining (ASYNCcnt).
//   B=64 graphs, NPG=2048 nodes/graph, C=64 in-ch, H=4 heads, D=64 head-dim
// ---------------------------------------------------------------------------

#define NB   64      // graphs
#define NPG  2048    // nodes per graph
#define CH   64      // in channels (= value dim)
#define NH   4       // heads
#define DH   64      // per-head dim
#define HD   (NH*DH) // 256
#define NN   (NB*NPG)
#define TN   64      // node tile
#define NT   (NPG/TN)

typedef __attribute__((ext_vector_type(16))) _Float16 v16h;
typedef __attribute__((ext_vector_type(8)))  _Float16 v8h;
typedef __attribute__((ext_vector_type(8)))  float    v8f;

// ---- WMMA helpers ----------------------------------------------------------
// A/B fragment (16-bit, 16x32): lane L -> row/col = L&15, kg = L>>4
//   element j(0..7)  -> K = k0 + kg*8 + j
//   element j(8..15) -> K = k0 + 16 + kg*8 + (j-8)
// Row-major [row][K] LDS source => two aligned 16-byte LDS loads per fragment.
__device__ __forceinline__ v16h frag_ld(const _Float16* __restrict__ base,
                                        int row, int stride, int k0, int kg) {
  const _Float16* p = base + row * stride + k0 + kg * 8;
  v8h lo = *(const v8h*)(p);
  v8h hi = *(const v8h*)(p + 16);
  return __builtin_shufflevector(lo, hi, 0,1,2,3,4,5,6,7,8,9,10,11,12,13,14,15);
}

__device__ __forceinline__ v8f wmma_f16(v16h a, v16h b, v8f c) {
  // D = A(16x32 f16) * B(32x16 f16) + C(16x16 f32)
  return __builtin_amdgcn_wmma_f32_16x16x32_f16(
      /*neg_a=*/false, a, /*neg_b=*/false, b,
      /*c_mod=*/(short)0, c, /*reuse_a=*/false, /*reuse_b=*/false);
}

// ---- CDNA5 async global->LDS copy (tracked by ASYNCcnt) --------------------
// Per ISA aperture rule, low 32 bits of a generic LDS pointer are the LDS addr.
__device__ __forceinline__ uint32_t lds_addr32(const void* p) {
  return (uint32_t)(uintptr_t)p;
}
__device__ __forceinline__ void async_copy_b128(uint32_t lds_off,
                                                const _Float16* gp) {
  asm volatile("global_load_async_to_lds_b128 %0, %1, off"
               :: "v"(lds_off), "v"((uint64_t)(uintptr_t)gp)
               : "memory");
}
__device__ __forceinline__ void wait_async_le2() {
  asm volatile("s_wait_asynccnt 0x2" ::: "memory");
}
__device__ __forceinline__ void wait_async_0() {
  asm volatile("s_wait_asynccnt 0x0" ::: "memory");
}

// ---- fp32 -> fp16 conversion (x, Wq, Wk) -----------------------------------
__global__ __launch_bounds__(256) void cvt_kernel(const float* __restrict__ s,
                                                  _Float16* __restrict__ d, int n4) {
  int i = blockIdx.x * blockDim.x + threadIdx.x;
  int stride = gridDim.x * blockDim.x;
  for (; i < n4; i += stride) {
    float4 v = ((const float4*)s)[i];
    _Float16 o[4] = { (_Float16)v.x, (_Float16)v.y, (_Float16)v.z, (_Float16)v.w };
    *(uint32_t*)(d + i * 4)     = *(uint32_t*)&o[0];
    *(uint32_t*)(d + i * 4 + 2) = *(uint32_t*)&o[2];
  }
}

// ---- Kernel A: per-(graph,head) stats: kv^T, k_sum, v_sum ------------------
// block = (b,h); 256 threads = 8 waves. kv[d][c] = sum_n kn[n][d]*x[n][c].
// x tile loads are double-buffered via global_load_async_to_lds_b128.
__global__ __launch_bounds__(256) void stats_kernel(
    const _Float16* __restrict__ xh,   // [N][CH] fp16
    const _Float16* __restrict__ wkh,  // [HD][CH] fp16 (row-major, as Wk_w)
    const float*    __restrict__ wkb,  // [HD]
    _Float16*       __restrict__ kvT,  // [B][H][CH c][DH d] fp16  (kv transposed)
    float*          __restrict__ ksum, // [B][H][DH]
    float*          __restrict__ vsum) // [B][CH]
{
  __shared__ _Float16 lds_x [2 * TN * CH]; // double-buffered [n][c]
  __shared__ _Float16 lds_xT[CH * TN];     // [c][n]
  __shared__ float    lds_kp[TN * DH];     // k-projection, fp32 [n][d]
  __shared__ _Float16 lds_knT[DH * TN];    // normalized k, [d][n]
  __shared__ _Float16 lds_w [DH * CH];     // Wk head, [d][c]
  __shared__ float    lds_b [DH];

  const int b = blockIdx.x >> 2;
  const int h = blockIdx.x & 3;
  const int tid  = threadIdx.x;
  const int wave = tid >> 5;
  const int lane = tid & 31;
  const int lrow = lane & 15;
  const int kg   = lane >> 4;

  // head weights + bias
  for (int i = tid; i < DH * CH / 8; i += 256)
    *(uint4*)(lds_w + i * 8) = *(const uint4*)(wkh + (size_t)h * DH * CH + i * 8);
  if (tid < DH) lds_b[tid] = wkb[h * DH + tid];

  // kv accumulators: 2 tiles per wave
  const int td  = wave >> 1;        // d-tile 0..3
  const int tc0 = (wave & 1) * 2;   // c-tile base 0 or 2
  v8f acc0 = {}; v8f acc1 = {};
  float ks_acc = 0.f, vs_acc = 0.f;

  const _Float16* xg = xh + (size_t)b * NPG * CH;
  const uint32_t  xbase = lds_addr32(lds_x);

  // prologue: async-copy tile 0 into buffer 0 (2 x b128 chunks per thread)
  async_copy_b128(xbase + tid * 16,         xg + tid * 8);
  async_copy_b128(xbase + (tid + 256) * 16, xg + (tid + 256) * 8);

  for (int t = 0; t < NT; ++t) {
    const int cur = t & 1;
    // issue next tile's async copy before waiting on the current one
    if (t + 1 < NT) {
      const _Float16* gn = xg + (size_t)(t + 1) * TN * CH;
      const uint32_t  lb = xbase + (uint32_t)(((t + 1) & 1) * (TN * CH * 2));
      async_copy_b128(lb + tid * 16,         gn + tid * 8);
      async_copy_b128(lb + (tid + 256) * 16, gn + (tid + 256) * 8);
      wait_async_le2();    // current tile's 2 chunks landed; next 2 in flight
    } else {
      wait_async_0();
    }
    __syncthreads();       // publish all waves' async copies; GEMM2(t-1) done

    const _Float16* xcur = lds_x + cur * (TN * CH);

    // build transposed copy [c][n] from the async-landed row-major tile
    for (int i = tid; i < TN * CH / 8; i += 256) {
      uint4 v = *(const uint4*)(xcur + i * 8);
      const _Float16* hp = (const _Float16*)&v;
      int node = (i * 8) >> 6, c0 = (i * 8) & 63;
      #pragma unroll
      for (int j = 0; j < 8; ++j) lds_xT[(c0 + j) * TN + node] = hp[j];
    }
    __syncthreads();

    // GEMM1: kproj = x_tile @ Wk_h^T   (16 tiles; 2 per wave)
    {
      const int rt = wave >> 1, cp = (wave & 1) * 2;
      #pragma unroll
      for (int cc = 0; cc < 2; ++cc) {
        const int ct = cp + cc;
        v8f acc = {};
        #pragma unroll
        for (int kk = 0; kk < CH; kk += 32) {
          v16h a  = frag_ld(xcur,  rt * 16 + lrow, CH, kk, kg);
          v16h bm = frag_ld(lds_w, ct * 16 + lrow, CH, kk, kg);
          acc = wmma_f16(a, bm, acc);
        }
        #pragma unroll
        for (int r = 0; r < 8; ++r)
          lds_kp[(rt * 16 + r + 8 * kg) * DH + ct * 16 + lrow] = acc[r];
      }
    }
    __syncthreads();

    // normalize (fp32): kn = (kproj + bias) / ||.||, stored transposed fp16
    if (tid < TN) {
      float ss = 0.f;
      for (int d = 0; d < DH; ++d) {
        float v = lds_kp[tid * DH + d] + lds_b[d];
        ss += v * v;
      }
      float inv = rsqrtf(ss);
      for (int d = 0; d < DH; ++d) {
        float v = (lds_kp[tid * DH + d] + lds_b[d]) * inv;
        lds_knT[d * TN + tid] = (_Float16)v;
      }
    }
    __syncthreads();

    // partial column sums (fp32 accumulate across the whole graph)
    if (tid < DH) {
      float s = 0.f;
      for (int n = 0; n < TN; ++n) s += (float)lds_knT[tid * TN + n];
      ks_acc += s;
    } else if (tid < DH + CH) {
      const int c = tid - DH;
      float s = 0.f;
      for (int n = 0; n < TN; ++n) s += (float)lds_xT[c * TN + n];
      vs_acc += s;
    }

    // GEMM2: kv += kn^T @ x_tile  (accumulated in registers, 2 tiles/wave)
    #pragma unroll
    for (int ks = 0; ks < TN; ks += 32) {
      v16h a  = frag_ld(lds_knT, td * 16 + lrow, TN, ks, kg);
      v16h b0 = frag_ld(lds_xT, (tc0 + 0) * 16 + lrow, TN, ks, kg);
      acc0 = wmma_f16(a, b0, acc0);
      v16h b1 = frag_ld(lds_xT, (tc0 + 1) * 16 + lrow, TN, ks, kg);
      acc1 = wmma_f16(a, b1, acc1);
    }
  }

  // write kv^T (fp16, [c][d] so kernel B's B-fragments are contiguous)
  _Float16* kvg = kvT + (size_t)(b * NH + h) * DH * CH;
  const int d0 = td * 16 + 8 * kg;  // 8 consecutive d per lane
  {
    const int c = (tc0 + 0) * 16 + lrow;
    _Float16 tmp[8];
    #pragma unroll
    for (int r = 0; r < 8; ++r) tmp[r] = (_Float16)acc0[r];
    *(uint4*)(kvg + c * DH + d0) = *(uint4*)tmp;
  }
  {
    const int c = (tc0 + 1) * 16 + lrow;
    _Float16 tmp[8];
    #pragma unroll
    for (int r = 0; r < 8; ++r) tmp[r] = (_Float16)acc1[r];
    *(uint4*)(kvg + c * DH + d0) = *(uint4*)tmp;
  }

  if (tid < DH)                          ksum[(b * NH + h) * DH + tid] = ks_acc;
  else if (tid < DH + CH && h == 0)      vsum[b * CH + (tid - DH)]     = vs_acc;
}

// ---- Kernel B: per-node output ---------------------------------------------
// block = (b, node-tile of 64); out[n, h*64+c] = (qn.kv + v_sum) / (qn.k_sum + n)
__global__ __launch_bounds__(256) void out_kernel(
    const _Float16* __restrict__ xh,
    const _Float16* __restrict__ wqh,
    const float*    __restrict__ wqb,
    const _Float16* __restrict__ kvT,
    const float*    __restrict__ ksum,
    const float*    __restrict__ vsum,
    const int*      __restrict__ n_nodes,
    float*          __restrict__ out)
{
  __shared__ _Float16 lds_x [TN * CH];   // [n][c]
  __shared__ float    lds_qp[TN * DH];   // q-projection fp32
  __shared__ _Float16 lds_qn[TN * DH];   // normalized q, row-major [n][d]
  __shared__ _Float16 lds_w [DH * CH];   // Wq head
  __shared__ _Float16 lds_kv[CH * DH];   // kv^T [c][d]
  __shared__ float    lds_b [DH];
  __shared__ float    lds_ks[DH];
  __shared__ float    lds_vs[CH];
  __shared__ float    lds_den[TN];

  const int b = blockIdx.x >> 5;           // / (NPG/TN)
  const int t = blockIdx.x & 31;
  const int tid  = threadIdx.x;
  const int wave = tid >> 5;
  const int lane = tid & 31;
  const int lrow = lane & 15;
  const int kg   = lane >> 4;

  const _Float16* xg = xh + ((size_t)b * NPG + (size_t)t * TN) * CH;
  for (int i = tid; i < TN * CH / 8; i += 256)
    *(uint4*)(lds_x + i * 8) = *(const uint4*)(xg + i * 8);
  if (tid < CH) lds_vs[tid] = vsum[b * CH + tid];
  const float nb = (float)n_nodes[b];

  for (int h = 0; h < NH; ++h) {
    __syncthreads();  // previous head done reading; lds_x visible (h==0)

    for (int i = tid; i < DH * CH / 8; i += 256) {
      *(uint4*)(lds_w  + i * 8) = *(const uint4*)(wqh + (size_t)h * DH * CH + i * 8);
      *(uint4*)(lds_kv + i * 8) = *(const uint4*)(kvT + (size_t)(b * NH + h) * DH * CH + i * 8);
    }
    if (tid < DH) {
      lds_b [tid] = wqb [h * DH + tid];
      lds_ks[tid] = ksum[(b * NH + h) * DH + tid];
    }
    __syncthreads();

    // GEMM1: qproj = x_tile @ Wq_h^T
    {
      const int rt = wave >> 1, cp = (wave & 1) * 2;
      #pragma unroll
      for (int cc = 0; cc < 2; ++cc) {
        const int ct = cp + cc;
        v8f acc = {};
        #pragma unroll
        for (int kk = 0; kk < CH; kk += 32) {
          v16h a  = frag_ld(lds_x, rt * 16 + lrow, CH, kk, kg);
          v16h bm = frag_ld(lds_w, ct * 16 + lrow, CH, kk, kg);
          acc = wmma_f16(a, bm, acc);
        }
        #pragma unroll
        for (int r = 0; r < 8; ++r)
          lds_qp[(rt * 16 + r + 8 * kg) * DH + ct * 16 + lrow] = acc[r];
      }
    }
    __syncthreads();

    // normalize + denominator (fp32):  den = (q . k_sum)/||q|| + n_nodes
    if (tid < TN) {
      float ss = 0.f, dk = 0.f;
      for (int d = 0; d < DH; ++d) {
        float v = lds_qp[tid * DH + d] + lds_b[d];
        ss += v * v;
        dk += v * lds_ks[d];
      }
      float inv = rsqrtf(ss);
      lds_den[tid] = dk * inv + nb;
      for (int d = 0; d < DH; ++d) {
        float v = (lds_qp[tid * DH + d] + lds_b[d]) * inv;
        lds_qn[tid * DH + d] = (_Float16)v;
      }
    }
    __syncthreads();

    // GEMM3 + fused epilogue: out = (qn @ kv + v_sum) / den
    {
      const int rt = wave >> 1, cp = (wave & 1) * 2;
      #pragma unroll
      for (int cc = 0; cc < 2; ++cc) {
        const int ct = cp + cc;
        v8f acc = {};
        #pragma unroll
        for (int kk = 0; kk < DH; kk += 32) {
          v16h a  = frag_ld(lds_qn, rt * 16 + lrow, DH, kk, kg);
          v16h bm = frag_ld(lds_kv, ct * 16 + lrow, DH, kk, kg);
          acc = wmma_f16(a, bm, acc);
        }
        const int c  = ct * 16 + lrow;
        const float vs = lds_vs[c];
        #pragma unroll
        for (int r = 0; r < 8; ++r) {
          const int n = rt * 16 + r + 8 * kg;
          const size_t node = (size_t)b * NPG + (size_t)t * TN + n;
          out[node * HD + h * DH + c] = (acc[r] + vs) / lds_den[n];
        }
      }
    }
  }
}

// ---------------------------------------------------------------------------
extern "C" void kernel_launch(void* const* d_in, const int* in_sizes, int n_in,
                              void* d_out, int out_size, void* d_ws, size_t ws_size,
                              hipStream_t stream) {
  (void)in_sizes; (void)n_in; (void)out_size; (void)ws_size;
  const float* x   = (const float*)d_in[0];
  const float* wq  = (const float*)d_in[1];
  const float* wqb = (const float*)d_in[2];
  const float* wk  = (const float*)d_in[3];
  const float* wkb = (const float*)d_in[4];
  const int*   nn  = (const int*)d_in[5];
  float* out = (float*)d_out;

  char* ws = (char*)d_ws;
  size_t off = 0;
  _Float16* xh  = (_Float16*)(ws + off); off += (size_t)NN * CH * 2;          // 16 MB
  _Float16* wqh = (_Float16*)(ws + off); off += (size_t)HD * CH * 2;          // 32 KB
  _Float16* wkh = (_Float16*)(ws + off); off += (size_t)HD * CH * 2;          // 32 KB
  _Float16* kvT = (_Float16*)(ws + off); off += (size_t)NB * NH * DH * CH * 2;// 2 MB
  float*    ks  = (float*)(ws + off);    off += (size_t)NB * NH * DH * 4;     // 64 KB
  float*    vs  = (float*)(ws + off);    off += (size_t)NB * CH * 4;          // 16 KB

  cvt_kernel<<<2048, 256, 0, stream>>>(x,  xh,  NN * CH / 4);
  cvt_kernel<<<16,   256, 0, stream>>>(wq, wqh, HD * CH / 4);
  cvt_kernel<<<16,   256, 0, stream>>>(wk, wkh, HD * CH / 4);
  stats_kernel<<<NB * NH, 256, 0, stream>>>(xh, wkh, wkb, kvT, ks, vs);
  out_kernel<<<NB * (NPG / TN), 256, 0, stream>>>(xh, wqh, wqb, kvT, ks, vs, nn, out);
}
